// NCELoss_71305047049043
// MI455X (gfx1250) — matrix operand order repaired and checked by reading, not compiled, not run on hardware
//
#include <hip/hip_runtime.h>

typedef __attribute__((ext_vector_type(16))) __bf16 v16bf;
typedef __attribute__((ext_vector_type(8)))  float  v8f;

#define NROW 4096
#define DDIM 256
#define GROWS 8192            // 2N
#define NCHUNK 8              // column chunks of 1024
#define LOG2E 1.44269504088896340736f
#define LN2   0.69314718055994530942f

// ---------------------------------------------------------------------------
// Kernel 1: normalize rows of [z1; z2] (f32) and split into bf16 hi/lo.
// One block (256 threads) per row; D == 256 so one element per thread.
// ---------------------------------------------------------------------------
__global__ __launch_bounds__(256) void prep_kernel(
    const float* __restrict__ x1, const float* __restrict__ x2,
    __bf16* __restrict__ ghi, __bf16* __restrict__ glo) {
  int row = blockIdx.x;                         // 0 .. 8191
  int t   = threadIdx.x;                        // 0 .. 255
  const float* src = (row < NROW) ? (x1 + (size_t)row * DDIM)
                                  : (x2 + (size_t)(row - NROW) * DDIM);
  float v = src[t];

  __shared__ float red[256];
  red[t] = v * v;
  __syncthreads();
#pragma unroll
  for (int s = 128; s > 0; s >>= 1) {
    if (t < s) red[t] += red[t + s];
    __syncthreads();
  }
  float nrm = sqrtf(red[0]);
  float rn  = 1.0f / fmaxf(nrm, 1e-8f);         // eps clamp as in reference

  float z   = v * rn;
  __bf16 h  = (__bf16)z;                        // RNE
  __bf16 l  = (__bf16)(z - (float)h);           // residual
  ghi[(size_t)row * DDIM + t] = h;
  glo[(size_t)row * DDIM + t] = l;
}

// ---------------------------------------------------------------------------
// Kernel 2: tiled WMMA pass. Each wave handles one (16-row tile, 1024-col
// chunk) task: A panel (16x256, hi+lo) resident in VGPRs, streams 64 column
// tiles, 3 bf16 WMMAs per K=32 chunk (hi*hi + hi*lo + lo*hi ~= fp32) with
// three independent accumulator chains for ILP.
// Fixed logsumexp shift of 10 (max possible logit) -> pure streaming sum.
// Partials stored per (row, chunk) -> deterministic combine in kernel 3.
// ---------------------------------------------------------------------------
__global__ __launch_bounds__(256) void nce_main_kernel(
    const __bf16* __restrict__ ghi, const __bf16* __restrict__ glo,
    float* __restrict__ sum2d, float* __restrict__ pv2d) {
  const int wave = threadIdx.x >> 5;
  const int lane = threadIdx.x & 31;
  const int task = blockIdx.x * 8 + wave;       // 4096 tasks total
  const int chunk   = task & 7;                 // 8 column chunks of 1024
  const int rowTile = task >> 3;                // 512 row tiles of 16
  const int rowBase = rowTile * 16;
  const int colBase = chunk * 1024;

  const int m  = lane & 15;                     // row/col within tile
  const int kh = lane >> 4;                     // K half selector

  // ---- Load resident A panel: 16 rows x 256, hi and lo -------------------
  // 16-bit A 16x32 layout: lanes 0-15 hold M=m with K = 0..7 (v0-3) and
  // 16..23 (v4-7); lanes 16-31 hold K = 8..15 and 24..31.
  v16bf ahi[8], alo[8];
  {
    const __bf16* arh = ghi + (size_t)(rowBase + m) * DDIM;
    const __bf16* arl = glo + (size_t)(rowBase + m) * DDIM;
#pragma unroll
    for (int k = 0; k < 8; ++k) {
      int off = k * 32 + kh * 8;
      v16bf a, b;
      ((uint4*)&a)[0] = *(const uint4*)(arh + off);
      ((uint4*)&a)[1] = *(const uint4*)(arh + off + 16);
      ((uint4*)&b)[0] = *(const uint4*)(arl + off);
      ((uint4*)&b)[1] = *(const uint4*)(arl + off + 16);
      ahi[k] = a;
      alo[k] = b;
    }
  }

  float accs[8];                                // per-row exp sums (partial)
  float pvr[8];                                 // positive-pair logit capture
#pragma unroll
  for (int v = 0; v < 8; ++v) { accs[v] = 0.0f; pvr[v] = 0.0f; }

  for (int ct = 0; ct < 64; ++ct) {
    const int colT = colBase + ct * 16;
    // B 32x16 layout: lane holds column n = m from source row (colT+m);
    // lanes 0-15 K=0..7,16..23, lanes 16-31 K=8..15,24..31 (matches A).
    const __bf16* brh = ghi + (size_t)(colT + m) * DDIM + kh * 8;
    const __bf16* brl = glo + (size_t)(colT + m) * DDIM + kh * 8;

    v8f c0 = {}; v8f c1 = {}; v8f c2 = {};      // 3 independent chains
#pragma unroll
    for (int k = 0; k < 8; ++k) {
      v16bf bh, bl;
      ((uint4*)&bh)[0] = *(const uint4*)(brh + k * 32);
      ((uint4*)&bh)[1] = *(const uint4*)(brh + k * 32 + 16);
      ((uint4*)&bl)[0] = *(const uint4*)(brl + k * 32);
      ((uint4*)&bl)[1] = *(const uint4*)(brl + k * 32 + 16);
      c0 = __builtin_amdgcn_wmma_f32_16x16x32_bf16(
          false, ahi[k], false, bh, (short)0, c0, false, false);
      c1 = __builtin_amdgcn_wmma_f32_16x16x32_bf16(
          false, ahi[k], false, bl, (short)0, c1, false, false);
      c2 = __builtin_amdgcn_wmma_f32_16x16x32_bf16(
          false, alo[k], false, bh, (short)0, c2, false, false);
    }

    // Epilogue: C/D layout -> (VGPR v, lane l): M = v + 8*(l>>4), N = l&15.
    const int gc = colT + m;
#pragma unroll
    for (int v = 0; v < 8; ++v) {
      float s  = (c0[v] + c1[v] + c2[v]) * 10.0f;  // logits = cos / T
      int   gr = rowBase + v + kh * 8;
      float e  = (gc == gr) ? 0.0f             // exclude self (diag of 11/22)
               : __builtin_amdgcn_exp2f((s - 10.0f) * LOG2E);
      accs[v] += e;
      if (gc == (gr ^ NROW)) pvr[v] = s;        // positive pair column
    }
  }

  // Reduce across the 16 lanes of each half-wave (covers all 16 columns of
  // a row), then store this chunk's partials (no atomics -> deterministic).
#pragma unroll
  for (int v = 0; v < 8; ++v) {
    float a = accs[v];
    float p = pvr[v];
#pragma unroll
    for (int msk = 8; msk >= 1; msk >>= 1) {
      a += __shfl_xor(a, msk, 32);
      p += __shfl_xor(p, msk, 32);              // at most one lane nonzero
    }
    if (m == 0) {
      int gr = rowBase + v + kh * 8;
      sum2d[gr * NCHUNK + chunk] = a;
      pv2d [gr * NCHUNK + chunk] = p;
    }
  }
}

// ---------------------------------------------------------------------------
// Kernel 3: deterministic combine.
//   loss[r] = log(sum_c sum2d[r][c]) + 10 - pv2d[r][chunk_of_partner]
// Partner column of row r is r^4096, so its chunk is (r^4096) >> 10.
// ---------------------------------------------------------------------------
__global__ __launch_bounds__(256) void finalize_kernel(
    const float* __restrict__ sum2d, const float* __restrict__ pv2d,
    float* __restrict__ out) {
  int r = blockIdx.x * 256 + threadIdx.x;
  if (r < GROWS) {
    float s = 0.0f;
#pragma unroll
    for (int c = 0; c < NCHUNK; ++c) s += sum2d[r * NCHUNK + c];
    float pvv = pv2d[r * NCHUNK + ((r ^ NROW) >> 10)];
    out[r] = __builtin_amdgcn_logf(s) * LN2 + 10.0f - pvv;
  }
}

extern "C" void kernel_launch(void* const* d_in, const int* in_sizes, int n_in,
                              void* d_out, int out_size, void* d_ws, size_t ws_size,
                              hipStream_t stream) {
  (void)in_sizes; (void)n_in; (void)out_size; (void)ws_size;
  const float* x1 = (const float*)d_in[0];
  const float* x2 = (const float*)d_in[1];
  float* out = (float*)d_out;

  char* ws = (char*)d_ws;
  const size_t gbytes = (size_t)GROWS * DDIM * sizeof(__bf16);   // 4 MB
  __bf16* ghi  = (__bf16*)ws;
  __bf16* glo  = (__bf16*)(ws + gbytes);
  float*  sum2d = (float*)(ws + 2 * gbytes);                     // 256 KB
  float*  pv2d  = sum2d + (size_t)GROWS * NCHUNK;                // 256 KB

  prep_kernel<<<GROWS, 256, 0, stream>>>(x1, x2, ghi, glo);
  nce_main_kernel<<<512, 256, 0, stream>>>(ghi, glo, sum2d, pv2d);
  finalize_kernel<<<GROWS / 256, 256, 0, stream>>>(sum2d, pv2d, out);
}
